// MyModel_15676630630737
// MI455X (gfx1250) — matrix operand order repaired and checked by reading, compiled, not run on hardware
//
#include <hip/hip_runtime.h>
#include <math.h>

#define T_ 512
#define B_ 64
#define E_ 128
#define H_ 256
#define K_ 9

typedef __attribute__((ext_vector_type(16))) __bf16 v16bf;
typedef __attribute__((ext_vector_type(8)))  float  v8f;

union Frag { v16bf v; uint4 q[2]; };

__device__ __forceinline__ unsigned short f2bf(float f) {
    union { float f; unsigned u; } c; c.f = f;
    unsigned u = c.u;
    unsigned r = u + 0x7FFFu + ((u >> 16) & 1u);   // round-to-nearest-even
    return (unsigned short)(r >> 16);
}
__device__ __forceinline__ float sigf(float x) { return 1.f / (1.f + expf(-x)); }

// A-fragment (16-bit A layout: per lane, K runs [8*hi,8*hi+8) and [16+8*hi,+8))
__device__ __forceinline__ void ldA(Frag& a, const unsigned short* base, int stride,
                                    int kc, int m, int hi) {
    const unsigned short* ab = base + m * stride + kc * 32;
    a.q[0] = *(const uint4*)(ab + 8 * hi);
    a.q[1] = *(const uint4*)(ab + 16 + 8 * hi);
}
// 4 B-fragments (gates i,f,g,o) for this wave's column slice, from packed weights
__device__ __forceinline__ void ldB4(Frag* b, const unsigned short* P, int nKC,
                                     int kc, int w, int lane) {
#pragma unroll
    for (int g = 0; g < 4; ++g) {
        const unsigned short* bp = P + ((size_t)((g * 16 + w) * nKC + kc) * 32 + lane) * 16;
        b[g].q[0] = *(const uint4*)bp;
        b[g].q[1] = *(const uint4*)(bp + 8);
    }
}
__device__ __forceinline__ void wmma4(v8f* acc, const Frag& a, const Frag* b) {
#pragma unroll
    for (int g = 0; g < 4; ++g)
        acc[g] = __builtin_amdgcn_wmma_f32_16x16x32_bf16(
            false, a.v, false, b[g].v, (short)0, acc[g], false, false);
}

// ---------------------------------------------------------------------------
// Pack an fp32 weight matrix (Krows x rowLen, row-major) into bf16 WMMA
// B-fragment order: elem[((nt*nKC+kc)*32+lane)*16+e] = W[kc*32+(e<8 ? 8*hi+e : 16+8*hi+e-8)][nt*16+n]
// ---------------------------------------------------------------------------
__global__ void pack_kernel(const float* __restrict__ src, unsigned short* __restrict__ dst,
                            int nNT, int nKC, int rowLen, int validCols) {
    int tid = blockIdx.x * blockDim.x + threadIdx.x;
    int total = nNT * nKC * 32;
    if (tid >= total) return;
    int lane = tid & 31;
    int kc   = (tid >> 5) % nKC;
    int nt   = tid / (32 * nKC);
    int hi = lane >> 4, n = lane & 15;
    int col = nt * 16 + n;
#pragma unroll
    for (int e = 0; e < 16; ++e) {
        int k = kc * 32 + (e < 8 ? 8 * hi + e : 16 + 8 * hi + (e - 8));
        float v = (col < validCols) ? src[(size_t)k * rowLen + col] : 0.f;
        dst[(size_t)tid * 16 + e] = f2bf(v);
    }
}

// ---------------------------------------------------------------------------
__global__ void lens_kernel(const int* __restrict__ text, int* __restrict__ lens_i,
                            float* __restrict__ lens_f) {
    __shared__ int cnt;
    int b = blockIdx.x, tid = threadIdx.x;
    if (tid == 0) cnt = 0;
    __syncthreads();
    int c = 0;
    for (int t = tid; t < T_; t += 128) c += (text[b * T_ + t] != 0);
    atomicAdd(&cnt, c);
    __syncthreads();
    if (tid == 0) { lens_i[b] = cnt; lens_f[b] = (float)cnt; }
}

// ---------------------------------------------------------------------------
// BiLSTM: 8 blocks = {dir 0/1} x {4 batch tiles of 16 rows}; 512 threads.
// Wave w owns gate columns [16w,16w+16): N-tiles {w,16+w,32+w,48+w} so the
// i/f/g/o accumulators for the same columns live in one wave's registers.
// K-chunk loops are rolled (manual x2 unroll) so weight fragments stream from
// L2 each step instead of being hoisted into registers and spilled.
// ---------------------------------------------------------------------------
__global__ __launch_bounds__(512) void bilstm_kernel(
    const int* __restrict__ text, const float* __restrict__ emb,
    const unsigned short* __restrict__ WfP, const unsigned short* __restrict__ UfP,
    const unsigned short* __restrict__ WbP, const unsigned short* __restrict__ UbP,
    const float* __restrict__ bfv, const float* __restrict__ bbv,
    unsigned short* __restrict__ hcat) {
    const int dir   = blockIdx.x >> 2;
    const int bbase = (blockIdx.x & 3) * 16;
    const unsigned short* WP = dir ? WbP : WfP;
    const unsigned short* UP = dir ? UbP : UfP;
    const float* bias        = dir ? bbv : bfv;

    __shared__ __align__(16) unsigned short x_lds[16 * 136]; // 128 cols, padded rows
    __shared__ __align__(16) unsigned short h_lds[16 * 264]; // 256 cols, padded rows
    __shared__ float c_lds[16 * 260];                        // padded for bank spread
    __shared__ int tok[16];

    const int tid  = threadIdx.x;
    const int w    = tid >> 5;     // wave 0..15
    const int lane = tid & 31;
    const int m    = lane & 15;
    const int hi   = lane >> 4;
    const int jn   = w * 16 + m;   // column within a gate (0..255)

    const float bi = bias[jn], bff = bias[H_ + jn], bg = bias[2 * H_ + jn], bo = bias[3 * H_ + jn];

    for (int i = tid; i < 16 * 264; i += 512) h_lds[i] = 0;
    for (int i = tid; i < 16 * 260; i += 512) c_lds[i] = 0.f;
    __syncthreads();

    for (int s = 0; s < T_; ++s) {
        const int t = dir ? (T_ - 1 - s) : s;
        if (tid < 16) tok[tid] = text[(bbase + tid) * T_ + t];
        __syncthreads();
        for (int e = 0; e < 4; ++e) {        // gather 16x128 embedding tile -> bf16 LDS
            int idx = tid + e * 512;
            int r = idx >> 7, cc = idx & 127;
            x_lds[r * 136 + cc] = f2bf(emb[(size_t)tok[r] * E_ + cc]);
        }
        __syncthreads();

        v8f acc[4];
#pragma unroll
        for (int g = 0; g < 4; ++g) acc[g] = (v8f){0.f, 0.f, 0.f, 0.f, 0.f, 0.f, 0.f, 0.f};

        // x @ W : K chunks 0..3 (rolled loop, x2 manual unroll -> bounded live regs)
#pragma unroll 1
        for (int kk = 0; kk < 2; ++kk) {
            Frag a0, a1, b0[4], b1[4];
            ldA(a0, x_lds, 136, 2 * kk, m, hi);     ldB4(b0, WP, 4, 2 * kk, w, lane);
            ldA(a1, x_lds, 136, 2 * kk + 1, m, hi); ldB4(b1, WP, 4, 2 * kk + 1, w, lane);
            wmma4(acc, a0, b0);
            wmma4(acc, a1, b1);
        }
        // h @ U : K chunks 0..7
#pragma unroll 1
        for (int kk = 0; kk < 4; ++kk) {
            Frag a0, a1, b0[4], b1[4];
            ldA(a0, h_lds, 264, 2 * kk, m, hi);     ldB4(b0, UP, 8, 2 * kk, w, lane);
            ldA(a1, h_lds, 264, 2 * kk + 1, m, hi); ldB4(b1, UP, 8, 2 * kk + 1, w, lane);
            wmma4(acc, a0, b0);
            wmma4(acc, a1, b1);
        }
        __syncthreads();  // all waves done reading h_lds before rewrite

#pragma unroll
        for (int r = 0; r < 8; ++r) {        // C layout: reg r -> row r + 8*hi, col lane%16
            int row = r + 8 * hi;
            float zi = acc[0][r] + bi, zf = acc[1][r] + bff;
            float zg = acc[2][r] + bg, zo = acc[3][r] + bo;
            float c = sigf(zf) * c_lds[row * 260 + jn] + sigf(zi) * tanhf(zg);
            float h = sigf(zo) * tanhf(c);
            c_lds[row * 260 + jn] = c;
            unsigned short hb = f2bf(h);
            h_lds[row * 264 + jn] = hb;
            hcat[((size_t)(bbase + row) * T_ + t) * (2 * H_) + dir * H_ + jn] = hb;
        }
        __syncthreads();
    }
}

// ---------------------------------------------------------------------------
// logits = hcat(32768x512 bf16) @ Wd(512x9) + bd.  N padded to 16, WMMA K=32.
// 256 blocks x 8 waves; each wave one 16-row tile. Rolled K loop (x2 unroll).
// ---------------------------------------------------------------------------
__global__ __launch_bounds__(256) void logits_kernel(
    const unsigned short* __restrict__ hcat, const unsigned short* __restrict__ WdP,
    const float* __restrict__ bd, float* __restrict__ out) {
    const int tid = threadIdx.x;
    const int w = tid >> 5, lane = tid & 31, m = lane & 15, hi = lane >> 4;
    const int tile = blockIdx.x * 8 + w;
    v8f acc = (v8f){0.f, 0.f, 0.f, 0.f, 0.f, 0.f, 0.f, 0.f};
    const unsigned short* arow = hcat + (size_t)(tile * 16 + m) * (2 * H_);
#pragma unroll 1
    for (int kk = 0; kk < 8; ++kk) {
        Frag a0, a1, b0, b1;
        ldA(a0, arow, 0, 2 * kk, 0, hi);      // stride unused: single row, m folded in arow
        ldA(a1, arow, 0, 2 * kk + 1, 0, hi);
        const unsigned short* bp0 = WdP + ((size_t)(2 * kk) * 32 + lane) * 16;
        const unsigned short* bp1 = WdP + ((size_t)(2 * kk + 1) * 32 + lane) * 16;
        b0.q[0] = *(const uint4*)bp0; b0.q[1] = *(const uint4*)(bp0 + 8);
        b1.q[0] = *(const uint4*)bp1; b1.q[1] = *(const uint4*)(bp1 + 8);
        acc = __builtin_amdgcn_wmma_f32_16x16x32_bf16(false, a0.v, false, b0.v, (short)0, acc, false, false);
        acc = __builtin_amdgcn_wmma_f32_16x16x32_bf16(false, a1.v, false, b1.v, (short)0, acc, false, false);
    }
    if (m < K_) {
        float bb = bd[m];
#pragma unroll
        for (int r = 0; r < 8; ++r) {
            int row = tile * 16 + r + 8 * hi;
            out[(size_t)row * K_ + m] = acc[r] + bb;
        }
    }
}

// ---------------------------------------------------------------------------
// CRF log-likelihood: one wave per batch element; lanes 0..8 hold alpha[k].
// ---------------------------------------------------------------------------
__global__ __launch_bounds__(32) void crf_kernel(
    const float* __restrict__ logits, const int* __restrict__ labels,
    const int* __restrict__ lens, const float* __restrict__ trans,
    float* __restrict__ ll_out) {
    int b = blockIdx.x, lane = threadIdx.x;
    int len = lens[b];
    const float* lg = logits + (size_t)b * T_ * K_;
    const int*  lab = labels + (size_t)b * T_;

    float us = 0.f, bs = 0.f;
    for (int t = lane; t < len; t += 32) {
        us += lg[t * K_ + lab[t]];
        if (t >= 1) bs += trans[lab[t - 1] * K_ + lab[t]];
    }
    float tot = us + bs;
    for (int off = 16; off > 0; off >>= 1) tot += __shfl_down(tot, off, 32);
    tot = __shfl(tot, 0, 32);

    int kp = (lane < K_) ? lane : 0;
    float tcol[K_];
#pragma unroll
    for (int k = 0; k < K_; ++k) tcol[k] = trans[k * K_ + kp];
    float alpha = (lane < K_) ? lg[lane] : -1e30f;
    for (int t = 1; t < len; ++t) {
        float vals[K_], mx = -1e30f;
#pragma unroll
        for (int k = 0; k < K_; ++k) {
            float ak = __shfl(alpha, k, 32);
            vals[k] = ak + tcol[k];
            mx = fmaxf(mx, vals[k]);
        }
        float s = 0.f;
#pragma unroll
        for (int k = 0; k < K_; ++k) s += expf(vals[k] - mx);
        alpha = mx + logf(s) + lg[t * K_ + kp];
    }
    float mx = -1e30f;
#pragma unroll
    for (int k = 0; k < K_; ++k) mx = fmaxf(mx, __shfl(alpha, k, 32));
    float s = 0.f;
#pragma unroll
    for (int k = 0; k < K_; ++k) s += expf(__shfl(alpha, k, 32) - mx);
    if (lane == 0) ll_out[b] = tot - (mx + logf(s));
}

// ---------------------------------------------------------------------------
extern "C" void kernel_launch(void* const* d_in, const int* in_sizes, int n_in,
                              void* d_out, int out_size, void* d_ws, size_t ws_size,
                              hipStream_t stream) {
    const int*   text   = (const int*)d_in[0];
    const int*   labels = (const int*)d_in[1];
    const float* emb    = (const float*)d_in[2];
    const float* Wf     = (const float*)d_in[3];
    const float* Uf     = (const float*)d_in[4];
    const float* bfv    = (const float*)d_in[5];
    const float* Wb     = (const float*)d_in[6];
    const float* Ub     = (const float*)d_in[7];
    const float* bbv    = (const float*)d_in[8];
    const float* Wd     = (const float*)d_in[9];
    const float* bd     = (const float*)d_in[10];
    const float* trans  = (const float*)d_in[11];
    float* out = (float*)d_out;

    char* ws = (char*)d_ws;
    const size_t OFF_WF   = 256;
    const size_t OFF_UF   = OFF_WF + 262144;
    const size_t OFF_WB   = OFF_UF + 524288;
    const size_t OFF_UB   = OFF_WB + 262144;
    const size_t OFF_WD   = OFF_UB + 524288;
    const size_t OFF_HCAT = OFF_WD + 16384;
    int* lens_i = (int*)ws;
    unsigned short* WfP  = (unsigned short*)(ws + OFF_WF);
    unsigned short* UfP  = (unsigned short*)(ws + OFF_UF);
    unsigned short* WbP  = (unsigned short*)(ws + OFF_WB);
    unsigned short* UbP  = (unsigned short*)(ws + OFF_UB);
    unsigned short* WdP  = (unsigned short*)(ws + OFF_WD);
    unsigned short* hcat = (unsigned short*)(ws + OFF_HCAT);

    pack_kernel<<<(64 * 4 * 32 + 255) / 256, 256, 0, stream>>>(Wf, WfP, 64, 4, 4 * H_, 4 * H_);
    pack_kernel<<<(64 * 8 * 32 + 255) / 256, 256, 0, stream>>>(Uf, UfP, 64, 8, 4 * H_, 4 * H_);
    pack_kernel<<<(64 * 4 * 32 + 255) / 256, 256, 0, stream>>>(Wb, WbP, 64, 4, 4 * H_, 4 * H_);
    pack_kernel<<<(64 * 8 * 32 + 255) / 256, 256, 0, stream>>>(Ub, UbP, 64, 8, 4 * H_, 4 * H_);
    pack_kernel<<<(1 * 16 * 32 + 255) / 256, 256, 0, stream>>>(Wd, WdP, 1, 16, K_, K_);

    lens_kernel<<<B_, 128, 0, stream>>>(text, lens_i, out + B_ * T_ * K_);
    bilstm_kernel<<<8, 512, 0, stream>>>(text, emb, WfP, UfP, WbP, UbP, bfv, bbv, hcat);
    logits_kernel<<<256, 256, 0, stream>>>(hcat, WdP, bd, out);
    crf_kernel<<<B_, 32, 0, stream>>>(out, labels, lens_i, trans, out + B_ * T_ * K_ + B_);
}
